// VQVAE_Trans_43026982372008
// MI455X (gfx1250) — matrix operand order repaired and checked by reading, not compiled, not run on hardware
//
#include <hip/hip_runtime.h>
#include <hip/hip_bf16.h>

typedef __attribute__((ext_vector_type(16))) _Float16 v16h;
typedef __attribute__((ext_vector_type(8)))  float    v8f;

#define WMMA_F16(a, b, c) \
  __builtin_amdgcn_wmma_f32_16x16x32_f16(false, (a), false, (b), (short)0, (c), false, false)

#define TT 9
#define EE 32
#define ROWS 144   // 16 sequences * 9 tokens
#define TSEQ 16

struct BlockW {
  const float *ln1_g, *ln1_b, *wqkv, *bqkv, *wproj, *bproj;
  const float *ln2_g, *ln2_b, *wfc, *bfc, *wmp, *bmp;
};
struct KP {
  const float* x;
  const float *wte, *bte, *pos;
  BlockW blk[2];
  const float *lnf_g, *lnf_b, *head_w, *head_b, *codebook;
  const float *dw1, *db1, *dg1, *dbe1, *dw2, *db2, *dg2, *dbe2, *dw3, *db3;
  float* out;
  int B;
};

// ---- WMMA fragment helpers (contiguous k order K = 16*(lane/16) + i, wave32) ----

// A fragment (16x32 f16, M x K) from row-major LDS panel.
// lane%16 = M row; element i: K = i + (lane>=16 ? 16 : 0)
__device__ __forceinline__ v16h a_frag_lds(const _Float16* buf, int row0, int ld, int lane) {
  const int m = lane & 15;
  const int kb = (lane & 16) ? 16 : 0;
  const _Float16* pr = buf + (size_t)(row0 + m) * ld + kb;
  v16h a;
#pragma unroll
  for (int i = 0; i < 16; ++i) {
    const int k = i;
    a[i] = pr[k];
  }
  return a;
}

// B fragment (32x16 f16, K x N) from an LDS weight staged transposed as [Npad][Kpad].
// lane%16 = N col; lanes 0-15: K=k0..k0+15 ; lanes 16-31: K=k0+16..k0+31.
// Per lane this is 16 contiguous f16 -> two ds_load_b128.
__device__ __forceinline__ v16h b_frag_lds(const _Float16* buf, int k0, int n0, int Kpad, int lane) {
  const int n = n0 + (lane & 15);
  const _Float16* pr = buf + (size_t)n * Kpad + k0 + ((lane & 16) ? 16 : 0);
  v16h b;
#pragma unroll
  for (int i = 0; i < 16; ++i) b[i] = pr[i];
  return b;
}

// Stage a row-major f32 weight [K x ldn] into LDS f16 transposed [Npad][Kpad], zero padded.
__device__ __forceinline__ void stage_wt(_Float16* dst, const float* W, int K, int N, int ldn,
                                         int Kpad, int Npad, int lane) {
  for (int idx = lane; idx < Kpad * Npad; idx += 32) {
    const int n = idx / Kpad;
    const int k = idx - n * Kpad;
    float v = (k < K && n < N) ? W[(size_t)k * ldn + n] : 0.0f;
    dst[idx] = (_Float16)v;
  }
}

__device__ __forceinline__ float gelu_tanh(float x) {
  const float c = 0.7978845608028654f;
  float t = tanhf(c * (x + 0.044715f * x * x * x));
  return 0.5f * x * (1.0f + t);
}

// LayerNorm rows of an LDS f32 panel -> f16 LDS panel (optionally ELU after affine)
__device__ __forceinline__ void ln_rows(const float* src, int ld_s, _Float16* dst, int ld_d,
                                        int nrows, int width, const float* g, const float* bb,
                                        bool elu, int lane) {
  const float invw = 1.0f / (float)width;
  for (int r = lane; r < nrows; r += 32) {
    const float* row = src + (size_t)r * ld_s;
    float mu = 0.0f;
    for (int c = 0; c < width; ++c) mu += row[c];
    mu *= invw;
    float var = 0.0f;
    for (int c = 0; c < width; ++c) { float d = row[c] - mu; var += d * d; }
    var *= invw;
    const float rs = rsqrtf(var + 1e-5f);
    _Float16* drow = dst + (size_t)r * ld_d;
    for (int c = 0; c < width; ++c) {
      float y = (row[c] - mu) * rs * g[c] + bb[c];
      if (elu) y = (y > 0.0f) ? y : (__expf(y) - 1.0f);
      drow[c] = (_Float16)y;
    }
  }
}

__global__ __launch_bounds__(32) void vqvae_fused(KP p) {
  const int lane = threadIdx.x;         // one wave32 per workgroup: no barriers needed
  const int s0 = blockIdx.x * TSEQ;     // base sequence index
  const int nn = lane & 15;
  const int mhalf = (lane & 16) ? 8 : 0;

  // LDS carve-out: 64512 bytes total, with phase overlays
  __shared__ __align__(16) unsigned char smem[64512];
  float*     hbuf   = (float*)(smem + 0);            // [144][32] f32 residual (18KB)
  _Float16*  qkvbuf = (_Float16*)(smem + 18432);     // [144][96] f16 (27KB)
  _Float16*  xbuf   = (_Float16*)(smem + 18432);     // overlay: [144][64] f16 input tile
  _Float16*  mwfc   = (_Float16*)(smem + 18432);     // overlay (MLP): wfc^T [128][32] f16 (8KB)
  _Float16*  mwmp   = (_Float16*)(smem + 26624);     // overlay (MLP): wmp^T [32][128] f16 (8KB)
  float*     dstage = (float*)(smem + 18432);        // overlay (decoder): [16][128] f32 (8KB)
  _Float16*  dwbuf  = (_Float16*)(smem + 26624);     // overlay (decoder): weights (16KB)
  _Float16*  abuf   = (_Float16*)(smem + 46080);     // [144][32] f16 / decoder [16][128] (9KB)
  _Float16*  gbuf   = (_Float16*)(smem + 55296);     // [16][32] f16 MLP K-staging (1KB)
  _Float16*  wbuf   = (_Float16*)(smem + 56320);     // [3072] f16 weight staging (6KB)
  float*     zbuf   = (float*)(smem + 62464);        // [16][16] f32 latents (1KB)
  float*     qzbuf  = (float*)(smem + 63488);        // [16][16] f32 quantized (1KB)

  const size_t quant_off = (size_t)p.B * 45;
  const size_t z_off     = (size_t)p.B * 61;

  // ---------------- Embedding: h = x @ wte + bte + pos ----------------
  {
    // h pre-init with bte + pos
    for (int idx = lane; idx < ROWS * EE; idx += 32) {
      const int r = idx >> 5, c = idx & 31;
      hbuf[idx] = p.bte[c] + p.pos[(r % TT) * EE + c];
    }
    // input tile -> f16, K padded 45 -> 64
    const float* xg = p.x + (size_t)s0 * TT * 45;    // contiguous [144][45]
    for (int idx = lane; idx < ROWS * 64; idx += 32) {
      const int r = idx >> 6, c = idx & 63;
      xbuf[idx] = (_Float16)((c < 45) ? xg[r * 45 + c] : 0.0f);
    }
    stage_wt(wbuf, p.wte, 45, EE, EE, 64, EE, lane); // wte^T [32][64]
    v16h wb00 = b_frag_lds(wbuf, 0, 0, 64, lane);
    v16h wb10 = b_frag_lds(wbuf, 32, 0, 64, lane);
    v16h wb01 = b_frag_lds(wbuf, 0, 16, 64, lane);
    v16h wb11 = b_frag_lds(wbuf, 32, 16, 64, lane);
    for (int m = 0; m < 9; ++m) {
      v16h a0 = a_frag_lds(xbuf, m * 16, 64, lane);
      v16h a1 = a_frag_lds(xbuf + 32, m * 16, 64, lane);
      v8f acc0 = {}, acc1 = {};
      acc0 = WMMA_F16(a0, wb00, acc0);
      acc0 = WMMA_F16(a1, wb10, acc0);
      acc1 = WMMA_F16(a0, wb01, acc1);
      acc1 = WMMA_F16(a1, wb11, acc1);
      const int mo = m * 16 + mhalf;
#pragma unroll
      for (int v = 0; v < 8; ++v) {
        hbuf[(mo + v) * EE + nn]      += acc0[v];
        hbuf[(mo + v) * EE + 16 + nn] += acc1[v];
      }
    }
  }

  // ---------------- Transformer blocks ----------------
  for (int l = 0; l < 2; ++l) {
    const BlockW& w = p.blk[l];

    // ln1 -> abuf (f16)
    ln_rows(hbuf, EE, abuf, EE, ROWS, EE, w.ln1_g, w.ln1_b, false, lane);

    // qkv = a @ wqkv + bqkv   [144,32]@[32,96] -> qkvbuf f16
    stage_wt(wbuf, w.wqkv, EE, 96, 96, 32, 96, lane); // wqkv^T [96][32] = 3072 f16
    {
      v16h qb[6];
      float qbias[6];
#pragma unroll
      for (int n = 0; n < 6; ++n) {
        qb[n] = b_frag_lds(wbuf, 0, n * 16, 32, lane);
        qbias[n] = w.bqkv[n * 16 + nn];
      }
      for (int m = 0; m < 9; ++m) {
        v16h a = a_frag_lds(abuf, m * 16, EE, lane);
        const int mo = m * 16 + mhalf;
#pragma unroll
        for (int n = 0; n < 6; ++n) {
          v8f acc = {};
          acc = WMMA_F16(a, qb[n], acc);
          const int col = n * 16 + nn;
#pragma unroll
          for (int v = 0; v < 8; ++v)
            qkvbuf[(mo + v) * 96 + col] = (_Float16)(acc[v] + qbias[n]);
        }
      }
    }

    // attention: 16 seq * 4 heads = 64 (s,h) pairs -> 2 per lane; T=9, HD=8
    const float scale = 0.35355339059327373f; // 1/sqrt(8)
#pragma unroll
    for (int pp = 0; pp < 2; ++pp) {
      const int idx = lane + 32 * pp;
      const int s = idx >> 2, hh = idx & 3;
      const _Float16* qb = qkvbuf + (s * 9) * 96 + hh * 8;
      const _Float16* kbp = qb + 32;
      const _Float16* vbp = qb + 64;
#pragma unroll
      for (int t = 0; t < 9; ++t) {
        float sc[9];
        float mx = -3.4e38f;
        for (int u = 0; u <= t; ++u) {
          float d = 0.0f;
#pragma unroll
          for (int e = 0; e < 8; ++e)
            d += (float)qb[t * 96 + e] * (float)kbp[u * 96 + e];
          d *= scale;
          sc[u] = d;
          mx = fmaxf(mx, d);
        }
        float sum = 0.0f;
        for (int u = 0; u <= t; ++u) { sc[u] = __expf(sc[u] - mx); sum += sc[u]; }
        const float inv = 1.0f / sum;
        float ov[8] = {0, 0, 0, 0, 0, 0, 0, 0};
        for (int u = 0; u <= t; ++u) {
          const float wt = sc[u] * inv;
#pragma unroll
          for (int e = 0; e < 8; ++e) ov[e] += wt * (float)vbp[u * 96 + e];
        }
        _Float16* orow = abuf + (s * 9 + t) * EE + hh * 8;
#pragma unroll
        for (int e = 0; e < 8; ++e) orow[e] = (_Float16)ov[e];
      }
    }

    // h += o @ wproj + bproj
    stage_wt(wbuf, w.wproj, EE, EE, EE, 32, 32, lane); // wproj^T [32][32]
    {
      v16h pb0 = b_frag_lds(wbuf, 0, 0, 32, lane);
      v16h pb1 = b_frag_lds(wbuf, 0, 16, 32, lane);
      const float pbias0 = w.bproj[nn];
      const float pbias1 = w.bproj[16 + nn];
      for (int m = 0; m < 9; ++m) {
        v16h a = a_frag_lds(abuf, m * 16, EE, lane);
        v8f acc0 = {}, acc1 = {};
        acc0 = WMMA_F16(a, pb0, acc0);
        acc1 = WMMA_F16(a, pb1, acc1);
        const int mo = m * 16 + mhalf;
#pragma unroll
        for (int v = 0; v < 8; ++v) {
          hbuf[(mo + v) * EE + nn]      += acc0[v] + pbias0;
          hbuf[(mo + v) * EE + 16 + nn] += acc1[v] + pbias1;
        }
      }
    }

    // ln2 -> abuf
    ln_rows(hbuf, EE, abuf, EE, ROWS, EE, w.ln2_g, w.ln2_b, false, lane);

    // MLP: fc (N=128) -> gelu -> mp (K=128), chunked through gbuf [16][32].
    // qkvbuf region is free now -> stage both MLP weights there.
    stage_wt(mwfc, w.wfc, EE, 128, 128, 32, 128, lane);  // wfc^T [128][32]
    stage_wt(mwmp, w.wmp, 128, EE, EE, 128, 32, lane);   // wmp^T [32][128]
    {
      float fbias[8];
#pragma unroll
      for (int j = 0; j < 8; ++j) fbias[j] = w.bfc[j * 16 + nn];
      const float mb0 = w.bmp[nn];
      const float mb1 = w.bmp[16 + nn];
      for (int m = 0; m < 9; ++m) {
        v16h a = a_frag_lds(abuf, m * 16, EE, lane);
        v8f acc0 = {}, acc1 = {};
        for (int jp = 0; jp < 4; ++jp) {
#pragma unroll
          for (int jj = 0; jj < 2; ++jj) {
            const int j = jp * 2 + jj;
            v8f g = {};
            v16h bf = b_frag_lds(mwfc, 0, j * 16, 32, lane);
            g = WMMA_F16(a, bf, g);
#pragma unroll
            for (int v = 0; v < 8; ++v)
              gbuf[(mhalf + v) * 32 + jj * 16 + nn] = (_Float16)gelu_tanh(g[v] + fbias[j]);
          }
          v16h a2  = a_frag_lds(gbuf, 0, 32, lane);
          v16h bm0 = b_frag_lds(mwmp, jp * 32, 0, 128, lane);
          v16h bm1 = b_frag_lds(mwmp, jp * 32, 16, 128, lane);
          acc0 = WMMA_F16(a2, bm0, acc0);
          acc1 = WMMA_F16(a2, bm1, acc1);
        }
        const int mo = m * 16 + mhalf;
#pragma unroll
        for (int v = 0; v < 8; ++v) {
          hbuf[(mo + v) * EE + nn]      += acc0[v] + mb0;
          hbuf[(mo + v) * EE + 16 + nn] += acc1[v] + mb1;
        }
      }
    }
  }

  // ---------------- lnf (last token only) + head -> z [16][16] ----------------
  if (lane < 16) {
    const float* row = hbuf + (lane * 9 + 8) * EE;
    float mu = 0.0f;
    for (int c = 0; c < EE; ++c) mu += row[c];
    mu *= (1.0f / EE);
    float var = 0.0f;
    for (int c = 0; c < EE; ++c) { float d = row[c] - mu; var += d * d; }
    var *= (1.0f / EE);
    const float rs = rsqrtf(var + 1e-5f);
    for (int c = 0; c < EE; ++c)
      abuf[lane * EE + c] = (_Float16)((row[c] - mu) * rs * p.lnf_g[c] + p.lnf_b[c]);
  }
  stage_wt(wbuf, p.head_w, EE, 16, 16, 32, 16, lane);   // head_w^T [16][32]
  {
    v16h a = a_frag_lds(abuf, 0, EE, lane);
    v16h b = b_frag_lds(wbuf, 0, 0, 32, lane);
    v8f z = {};
    z = WMMA_F16(a, b, z);
    const float hb = p.head_b[nn];
#pragma unroll
    for (int v = 0; v < 8; ++v) {
      const float zv = z[v] + hb;
      zbuf[(mhalf + v) * 16 + nn] = zv;
      p.out[z_off + (size_t)(s0 + mhalf + v) * 16 + nn] = zv;
    }
  }

  // ---------------- VQ: nearest codebook row (K=512, LAT=16) ----------------
  for (int s = 0; s < 16; ++s) {
    float zr[16];
#pragma unroll
    for (int d = 0; d < 16; ++d) zr[d] = zbuf[s * 16 + d];
    float bd = 3.4e38f;
    int bk = 0;
    for (int k = lane; k < 512; k += 32) {
      const float* cb = p.codebook + (size_t)k * 16;
      float dist = 0.0f;
#pragma unroll
      for (int d = 0; d < 16; ++d) { float df = zr[d] - cb[d]; dist += df * df; }
      if (dist < bd) { bd = dist; bk = k; }
    }
#pragma unroll
    for (int off = 16; off > 0; off >>= 1) {
      float od = __shfl_xor(bd, off, 32);
      int   ok = __shfl_xor(bk, off, 32);
      if (od < bd || (od == bd && ok < bk)) { bd = od; bk = ok; }
    }
    if (lane < 16) {
      const float qv = p.codebook[(size_t)bk * 16 + lane];
      qzbuf[s * 16 + lane] = qv;
      p.out[quant_off + (size_t)(s0 + s) * 16 + lane] = qv;
    }
  }

  // ---------------- Decoder: st == quantize (forward value) ----------------
  // stage A [16][32] f16 (K padded 16 -> 32)
  for (int idx = lane; idx < 16 * 32; idx += 32) {
    const int r = idx >> 5, c = idx & 31;
    abuf[idx] = (_Float16)((c < 16) ? qzbuf[r * 16 + c] : 0.0f);
  }
  // L1: [16,16]@[16,64] + b1
  stage_wt(dwbuf, p.dw1, 16, 64, 64, 32, 64, lane);     // dw1^T [64][32]
  {
    v16h a = a_frag_lds(abuf, 0, 32, lane);
#pragma unroll
    for (int n = 0; n < 4; ++n) {
      v8f acc = {};
      v16h b = b_frag_lds(dwbuf, 0, n * 16, 32, lane);
      acc = WMMA_F16(a, b, acc);
      const int col = n * 16 + nn;
      const float bias = p.db1[col];
#pragma unroll
      for (int v = 0; v < 8; ++v)
        dstage[(mhalf + v) * 64 + col] = acc[v] + bias;
    }
  }
  ln_rows(dstage, 64, abuf, 128, 16, 64, p.dg1, p.dbe1, true, lane);   // LN+ELU -> f16
  // L2: [16,64]@[64,128] + b2
  stage_wt(dwbuf, p.dw2, 64, 128, 128, 64, 128, lane);  // dw2^T [128][64]
  {
    v16h a0 = a_frag_lds(abuf, 0, 128, lane);
    v16h a1 = a_frag_lds(abuf + 32, 0, 128, lane);
    for (int n = 0; n < 8; ++n) {
      v8f acc = {};
      v16h b0 = b_frag_lds(dwbuf, 0, n * 16, 64, lane);
      v16h b1 = b_frag_lds(dwbuf, 32, n * 16, 64, lane);
      acc = WMMA_F16(a0, b0, acc);
      acc = WMMA_F16(a1, b1, acc);
      const int col = n * 16 + nn;
      const float bias = p.db2[col];
#pragma unroll
      for (int v = 0; v < 8; ++v)
        dstage[(mhalf + v) * 128 + col] = acc[v] + bias;
    }
  }
  ln_rows(dstage, 128, abuf, 128, 16, 128, p.dg2, p.dbe2, true, lane); // LN+ELU -> f16
  // L3: [16,128]@[128,45] + b3 -> input_hat
  stage_wt(dwbuf, p.dw3, 128, 45, 45, 128, 48, lane);   // dw3^T [48][128]
  {
    v16h a0 = a_frag_lds(abuf, 0, 128, lane);
    v16h a1 = a_frag_lds(abuf + 32, 0, 128, lane);
    v16h a2 = a_frag_lds(abuf + 64, 0, 128, lane);
    v16h a3 = a_frag_lds(abuf + 96, 0, 128, lane);
#pragma unroll
    for (int n = 0; n < 3; ++n) {
      v8f acc = {};
      v16h b0 = b_frag_lds(dwbuf, 0,  n * 16, 128, lane);
      v16h b1 = b_frag_lds(dwbuf, 32, n * 16, 128, lane);
      v16h b2 = b_frag_lds(dwbuf, 64, n * 16, 128, lane);
      v16h b3 = b_frag_lds(dwbuf, 96, n * 16, 128, lane);
      acc = WMMA_F16(a0, b0, acc);
      acc = WMMA_F16(a1, b1, acc);
      acc = WMMA_F16(a2, b2, acc);
      acc = WMMA_F16(a3, b3, acc);
      const int col = n * 16 + nn;
      if (col < 45) {
        const float bias = p.db3[col];
#pragma unroll
        for (int v = 0; v < 8; ++v)
          p.out[(size_t)(s0 + mhalf + v) * 45 + col] = acc[v] + bias;
      }
    }
  }
}

extern "C" void kernel_launch(void* const* d_in, const int* in_sizes, int n_in,
                              void* d_out, int out_size, void* d_ws, size_t ws_size,
                              hipStream_t stream) {
  (void)n_in; (void)out_size; (void)d_ws; (void)ws_size;
  KP kp;
  int i = 0;
  kp.x   = (const float*)d_in[i++];
  kp.wte = (const float*)d_in[i++];
  kp.bte = (const float*)d_in[i++];
  kp.pos = (const float*)d_in[i++];
  for (int l = 0; l < 2; ++l) {
    BlockW& b = kp.blk[l];
    b.ln1_g = (const float*)d_in[i++];
    b.ln1_b = (const float*)d_in[i++];
    b.wqkv  = (const float*)d_in[i++];
    b.bqkv  = (const float*)d_in[i++];
    b.wproj = (const float*)d_in[i++];
    b.bproj = (const float*)d_in[i++];
    b.ln2_g = (const float*)d_in[i++];
    b.ln2_b = (const float*)d_in[i++];
    b.wfc   = (const float*)d_in[i++];
    b.bfc   = (const float*)d_in[i++];
    b.wmp   = (const float*)d_in[i++];
    b.bmp   = (const float*)d_in[i++];
  }
  kp.lnf_g    = (const float*)d_in[i++];
  kp.lnf_b    = (const float*)d_in[i++];
  kp.head_w   = (const float*)d_in[i++];
  kp.head_b   = (const float*)d_in[i++];
  kp.codebook = (const float*)d_in[i++];
  kp.dw1  = (const float*)d_in[i++];
  kp.db1  = (const float*)d_in[i++];
  kp.dg1  = (const float*)d_in[i++];
  kp.dbe1 = (const float*)d_in[i++];
  kp.dw2  = (const float*)d_in[i++];
  kp.db2  = (const float*)d_in[i++];
  kp.dg2  = (const float*)d_in[i++];
  kp.dbe2 = (const float*)d_in[i++];
  kp.dw3  = (const float*)d_in[i++];
  kp.db3  = (const float*)d_in[i++];
  kp.out  = (float*)d_out;
  kp.B    = in_sizes[0] / (TT * 45);

  const int nblk = kp.B / TSEQ;  // 4096 workgroups of one wave32 each
  hipLaunchKernelGGL(vqvae_fused, dim3(nblk), dim3(32), 0, stream, kp);
}
